// TTT_44238163149123
// MI455X (gfx1250) — matrix operand order repaired and checked by reading, not compiled, not run on hardware
//
#include <hip/hip_runtime.h>
#include <hip/hip_bf16.h>
#include <stdint.h>

// ---------------------------------------------------------------------------
// TTT recurrence on MI455X (gfx1250):
//   per step t:  Xn = x_t + n_t
//                Z  = Xn @ W1^T ; H = gelu(Z)
//                P  = H  @ W2^T ; dP = 2/(B*D) * (P - x_t)
//                dH = dP @ W2   ; dZ = dH * gelu'(Z)
//                W1 -= LR * dZ^T @ Xn ;  W2 -= LR * dP^T @ H
//                out_t = gelu(x_t @ W1^T) @ W2^T
// Weights kept as f32 masters; bf16 (+ transposed bf16) copies feed WMMA.
// A-tiles staged to LDS with gfx1250 async copies (double-buffered, ASYNCcnt).
// ---------------------------------------------------------------------------

typedef __bf16 bf16;
typedef __attribute__((ext_vector_type(16))) __bf16 v16bf;
typedef __attribute__((ext_vector_type(8)))  __bf16 v8bf;
typedef __attribute__((ext_vector_type(8)))  float  v8f;

union Frag16 { v16bf v; v8bf h[2]; };

__device__ __forceinline__ float gelu_f(float x) {
    return 0.5f * x * (1.0f + erff(x * 0.70710678118654752f));
}
__device__ __forceinline__ float dgelu_f(float x) {
    return 0.5f * (1.0f + erff(x * 0.70710678118654752f))
         + x * 0.39894228040143268f * __expf(-0.5f * x * x);
}

// gfx1250 async global->LDS copy, 4 bytes per active lane (ASYNCcnt-tracked).
__device__ __forceinline__ void async_b32_to_lds(uint32_t lds_off, uint64_t gaddr) {
    asm volatile("global_load_async_to_lds_b32 %0, %1, off"
                 :: "v"(lds_off), "v"(gaddr) : "memory");
}
__device__ __forceinline__ void wait_async0() {
    asm volatile("s_wait_asynccnt 0x0" ::: "memory");
}

constexpr int EPI_F32   = 0;  // f32_out[m,n] = c
constexpr int EPI_GELU  = 1;  // f32_out = c ; b_out = gelu(c)
constexpr int EPI_DPRED = 2;  // b_out = scale*(c - f32_in)
constexpr int EPI_DGELU = 3;  // b_out = c * gelu'(f32_in)
constexpr int EPI_UPD   = 4;  // w = f32_out - LR*c ; f32_out = w ; b_out = w ; b_out2^T = w
constexpr int EPI_OUT   = 5;  // f32_out[m,n] = c   (strided into d_out)

// C[m,n] = sum_k A[m,k] * Bw[n,k]   (A row-major [M,K] bf16, Bw row-major [N,K] bf16)
// block = 256 threads (8 waves); each wave owns a 16x16 tile; block tile = 16 x 128.
template <int EPI>
__global__ __launch_bounds__(256) void gemm_bf16_k(
    const bf16* __restrict__ A,  int lda,
    const bf16* __restrict__ Bw, int ldb,
    int K,
    float* __restrict__ f32_out, int ld_fo,
    const float* __restrict__ f32_in, int ld_fi,
    bf16* __restrict__ b_out,  int ld_bo,
    bf16* __restrict__ b_out2, int ld_bo2)
{
    __shared__ bf16 As[2][16 * 32];              // double-buffered A tile

    const int tid  = threadIdx.x;
    const int wave = tid >> 5;
    const int lane = tid & 31;
    const int hl   = lane >> 4;                  // which 16-lane half
    const int l15  = lane & 15;

    const int m0 = blockIdx.y * 16;
    const int n0 = blockIdx.x * 128 + wave * 16;

    // cooperative A-tile copy mapping: each thread moves 2 contiguous bf16 (4 B)
    const int arow = tid >> 4;                   // 0..15
    const int acol = (tid & 15) * 2;             // 0,2,..,30
    const bf16* my_a = &A[(size_t)(m0 + arow) * lda + acol];
    const uint32_t my_lds = (uint32_t)(uintptr_t)(&As[0][0]) + (uint32_t)tid * 4u;

    v8f acc = {};

    const int nk = K >> 5;                       // # of K=32 chunks
    // prologue: kick off chunk 0 into buffer 0
    async_b32_to_lds(my_lds, (uint64_t)(uintptr_t)my_a);

    for (int ki = 0; ki < nk; ++ki) {
        const int b = ki & 1;
        wait_async0();                           // my chunk-ki bytes are in LDS
        __syncthreads();                         // everyone's are; prev reads done

        // overlap: start copying chunk ki+1 into the other buffer
        if (ki + 1 < nk)
            async_b32_to_lds(my_lds + (uint32_t)((b ^ 1) * 1024),
                             (uint64_t)(uintptr_t)(my_a + (ki + 1) * 32));

        const int k0 = ki * 32;

        // A fragment (16-bit A 16x32 layout): lanes 0-15 -> K 0..7 & 16..23,
        // lanes 16-31 -> K 8..15 & 24..31
        Frag16 af;
        const bf16* ap = &As[b][l15 * 32 + hl * 8];
        af.h[0] = *(const v8bf*)(ap);
        af.h[1] = *(const v8bf*)(ap + 16);

        // B fragment (16-bit B 32x16): lane holds column n0+l15; lanes 0-15 K 0..7 & 16..23,
        // lanes 16-31 K 8..15 & 24..31; Bw row n is contiguous in K.
        Frag16 bfr;
        const bf16* bp = &Bw[(size_t)(n0 + l15) * ldb + k0 + hl * 8];
        bfr.h[0] = *(const v8bf*)(bp);
        bfr.h[1] = *(const v8bf*)(bp + 16);

        if (ki + 1 < nk)                          // pull next B chunk into WGP$
            __builtin_prefetch(&Bw[(size_t)(n0 + l15) * ldb + k0 + 32], 0, 3);

        acc = __builtin_amdgcn_wmma_f32_16x16x32_bf16(
                  false, af.v, false, bfr.v, (short)0, acc, false, false);
    }

    // C/D layout: VGPR r -> M = m0 + hl*8 + r ; N = n0 + l15
    const int mb = m0 + hl * 8;
    const int n  = n0 + l15;
#pragma unroll
    for (int r = 0; r < 8; ++r) {
        const int m = mb + r;
        const float c = acc[r];
        if constexpr (EPI == EPI_F32) {
            f32_out[(size_t)m * ld_fo + n] = c;
        } else if constexpr (EPI == EPI_GELU) {
            f32_out[(size_t)m * ld_fo + n] = c;
            b_out[(size_t)m * ld_bo + n]   = (bf16)gelu_f(c);
        } else if constexpr (EPI == EPI_DPRED) {
            // dL/dpred = 2*(pred - x)/(B*D),  B*D = 65536
            const float d = 3.0517578125e-05f * (c - f32_in[(size_t)m * ld_fi + n]);
            b_out[(size_t)m * ld_bo + n] = (bf16)d;
        } else if constexpr (EPI == EPI_DGELU) {
            const float z = f32_in[(size_t)m * ld_fi + n];
            b_out[(size_t)m * ld_bo + n] = (bf16)(c * dgelu_f(z));
        } else if constexpr (EPI == EPI_UPD) {
            const size_t idx = (size_t)m * ld_fo + n;
            const float w = f32_out[idx] - 0.01f * c;   // LR = 0.01
            f32_out[idx] = w;
            b_out[(size_t)m * ld_bo + n] = (bf16)w;
            if (b_out2) b_out2[(size_t)n * ld_bo2 + m] = (bf16)w;
        } else if constexpr (EPI == EPI_OUT) {
            f32_out[(size_t)m * ld_fo + n] = c;
        }
    }
}

// Xn = x+n (bf16, row + transposed) and Xb = bf16(x)
__global__ __launch_bounds__(256) void prep_k(
    const float* __restrict__ x, const float* __restrict__ nz, int ldx,
    bf16* __restrict__ Xnb, bf16* __restrict__ Xb, bf16* __restrict__ XnTb,
    int B, int D)
{
    const int i = blockIdx.x * blockDim.x + threadIdx.x;   // 0 .. B*D-1
    const int b = i / D, d = i - b * D;
    const float xv = x[(size_t)b * ldx + d];
    const float nv = nz[(size_t)b * ldx + d];
    Xb[i] = (bf16)xv;
    const bf16 s = (bf16)(xv + nv);
    Xnb[i] = s;
    XnTb[(size_t)d * B + b] = s;
}

// transpose [B,D] -> [D,B] for H, dpred, dz (bf16)
__global__ __launch_bounds__(256) void tr3_k(
    const bf16* __restrict__ H, const bf16* __restrict__ dp, const bf16* __restrict__ dz,
    bf16* __restrict__ HT, bf16* __restrict__ dpT, bf16* __restrict__ dzT,
    int B, int D)
{
    const int i = blockIdx.x * blockDim.x + threadIdx.x;
    const int b = i / D, d = i - b * D;
    const size_t j = (size_t)d * B + b;
    HT[j]  = H[i];
    dpT[j] = dp[i];
    dzT[j] = dz[i];
}

// f32 masters + bf16 copies (+ transposed W2)
__global__ __launch_bounds__(256) void initw_k(
    const float* __restrict__ W1, const float* __restrict__ W2,
    float* __restrict__ W1m, float* __restrict__ W2m,
    bf16* __restrict__ W1b, bf16* __restrict__ W2b, bf16* __restrict__ W2Tb, int D)
{
    const int i = blockIdx.x * blockDim.x + threadIdx.x;   // 0 .. D*D-1
    const int r = i / D, c = i - r * D;
    const float w1 = W1[i], w2 = W2[i];
    W1m[i] = w1;  W2m[i] = w2;
    W1b[i] = (bf16)w1;
    W2b[i] = (bf16)w2;
    W2Tb[(size_t)c * D + r] = (bf16)w2;
}

extern "C" void kernel_launch(void* const* d_in, const int* in_sizes, int n_in,
                              void* d_out, int out_size, void* d_ws, size_t ws_size,
                              hipStream_t stream) {
    (void)in_sizes; (void)n_in; (void)out_size; (void)ws_size;
    constexpr int B = 64, S = 512, D = 1024;

    const float* x  = (const float*)d_in[0];
    const float* nz = (const float*)d_in[1];
    const float* W1 = (const float*)d_in[2];
    const float* W2 = (const float*)d_in[3];
    float* out = (float*)d_out;

    char* wsp = (char*)d_ws;
    auto alloc = [&](size_t bytes) -> char* {
        char* p = wsp; wsp += (bytes + 255) & ~(size_t)255; return p;
    };
    float* W1m  = (float*)alloc((size_t)D * D * 4);
    float* W2m  = (float*)alloc((size_t)D * D * 4);
    bf16*  W1b  = (bf16*) alloc((size_t)D * D * 2);
    bf16*  W2b  = (bf16*) alloc((size_t)D * D * 2);
    bf16*  W2Tb = (bf16*) alloc((size_t)D * D * 2);
    float* Zf   = (float*)alloc((size_t)B * D * 4);
    bf16*  Xnb  = (bf16*) alloc((size_t)B * D * 2);
    bf16*  Xb   = (bf16*) alloc((size_t)B * D * 2);
    bf16*  Hb   = (bf16*) alloc((size_t)B * D * 2);
    bf16*  Hob  = (bf16*) alloc((size_t)B * D * 2);
    bf16*  dPb  = (bf16*) alloc((size_t)B * D * 2);
    bf16*  dZb  = (bf16*) alloc((size_t)B * D * 2);
    bf16*  XnT  = (bf16*) alloc((size_t)B * D * 2);
    bf16*  HT   = (bf16*) alloc((size_t)B * D * 2);
    bf16*  dPT  = (bf16*) alloc((size_t)B * D * 2);
    bf16*  dZT  = (bf16*) alloc((size_t)B * D * 2);

    initw_k<<<(D * D) / 256, 256, 0, stream>>>(W1, W2, W1m, W2m, W1b, W2b, W2Tb, D);

    const dim3 gAct(D / 128, B / 16);   // activation GEMMs: M=64,  N=1024
    const dim3 gWgt(D / 128, D / 16);   // weight-grad GEMMs: M=1024, N=1024
    const int  eN = (B * D) / 256;      // elementwise grid

    for (int t = 0; t < S; ++t) {
        const float* xt = x  + (size_t)t * D;
        const float* nt = nz + (size_t)t * D;

        prep_k<<<eN, 256, 0, stream>>>(xt, nt, S * D, Xnb, Xb, XnT, B, D);

        // Z = Xn @ W1^T ; H = gelu(Z)
        gemm_bf16_k<EPI_GELU><<<gAct, 256, 0, stream>>>(
            Xnb, D, W1b, D, D, Zf, D, nullptr, 0, Hb, D, nullptr, 0);

        // P = H @ W2^T ; dP = scale*(P - x_t)
        gemm_bf16_k<EPI_DPRED><<<gAct, 256, 0, stream>>>(
            Hb, D, W2b, D, D, nullptr, 0, xt, S * D, dPb, D, nullptr, 0);

        // dH = dP @ W2 (= dP @ (W2^T)^T via W2Tb) ; dZ = dH * gelu'(Z)
        gemm_bf16_k<EPI_DGELU><<<gAct, 256, 0, stream>>>(
            dPb, D, W2Tb, D, D, nullptr, 0, Zf, D, dZb, D, nullptr, 0);

        tr3_k<<<eN, 256, 0, stream>>>(Hb, dPb, dZb, HT, dPT, dZT, B, D);

        // gW1 = dZ^T @ Xn ; W1 -= LR*gW1 ; refresh W1b
        gemm_bf16_k<EPI_UPD><<<gWgt, 256, 0, stream>>>(
            dZT, B, XnT, B, B, W1m, D, nullptr, 0, W1b, D, nullptr, 0);

        // gW2 = dP^T @ H ; W2 -= LR*gW2 ; refresh W2b and W2Tb
        gemm_bf16_k<EPI_UPD><<<gWgt, 256, 0, stream>>>(
            dPT, B, HT, B, B, W2m, D, nullptr, 0, W2b, D, W2Tb, D);

        // out_t = gelu(x_t @ W1'^T) @ W2'^T
        gemm_bf16_k<EPI_GELU><<<gAct, 256, 0, stream>>>(
            Xb, D, W1b, D, D, Zf, D, nullptr, 0, Hob, D, nullptr, 0);
        gemm_bf16_k<EPI_OUT><<<gAct, 256, 0, stream>>>(
            Hob, D, W2b, D, D, out + (size_t)t * D, S * D, nullptr, 0,
            nullptr, 0, nullptr, 0);
    }
}
